// AnchorStore_87935160418516
// MI455X (gfx1250) — matrix operand order repaired and checked by reading, not compiled, not run on hardware
//
#include <hip/hip_runtime.h>

// Problem constants (from reference)
#define D_DIM   50257            // feature dim
#define D_PAD   50272            // round up to multiple of 32 (WMMA K)
#define N_DBLK  1571             // D_PAD / 32
#define K_Q     8192             // queue size
#define B_Q     512              // batch of queries
#define LQ_BYTES ((size_t)D_PAD * B_Q * 2)   // bf16 swizzled log2(query): ~51.5 MB

typedef __attribute__((ext_vector_type(16))) __bf16        v16bf;
typedef __attribute__((ext_vector_type(8)))  float         v8f;
typedef __attribute__((ext_vector_type(4), aligned(4))) float f32x4; // anchor rows only DWORD-aligned (D odd)
typedef __attribute__((ext_vector_type(4)))  unsigned int  u32x4;

// Raw v_log_f32 (log2). Inputs are in (1e-6, 1): always normal, no fixups
// needed. Whole pipeline works in the log2 domain: kl2 = Sum a*log2(a) -
// Sum a*log2(q) = kl_nat / ln2 -- a positive rescale, rank-invariant for
// top-k, so ln2 is dropped everywhere.
__device__ __forceinline__ float fast_log2(float x) {
  return __builtin_amdgcn_logf(x);
}

// ---------------------------------------------------------------------------
// Kernel 1: lq = bf16(log2(query)), written pre-swizzled into WMMA B-fragment
// layout: fragment f = dblk*32 + bblk holds a 32(d) x 16(b) tile, 512 bf16,
// lane = h*16 + (b&15) owns 16 bf16 (d = 8h+s, 16+8h+s-8) at lane*32 bytes.
// Padding region d in [D_DIM, D_PAD) stores 0 (kills GEMM tail contribution).
// ---------------------------------------------------------------------------
__global__ void __launch_bounds__(256) logq_swizzle_kernel(
    const float* __restrict__ q, unsigned short* __restrict__ lq)
{
  const int d = blockIdx.x * 256 + threadIdx.x;
  const int b = blockIdx.y;
  if (d >= D_PAD) return;
  float v = 0.0f;
  if (d < D_DIM) v = fast_log2(q[(size_t)b * D_DIM + d]);
  const int dblk = d >> 5;
  const int h    = (d >> 3) & 1;   // lane half: 8-wide group within each 16-d half (table order)
  const int e    = (d & 7) | ((d >> 1) & 8);  // element within lane: k = 8h+s / 16+8h+(s-8)
  const int bblk = b >> 4;
  const int n    = b & 15;
  union { __bf16 hf; unsigned short s; } cv;
  cv.hf = (__bf16)v;
  lq[(((size_t)dblk * 32 + bblk) << 9) + (size_t)((h * 16 + n) << 4) + e] = cv.s;
}

// ---------------------------------------------------------------------------
// One K=32 step of the KL GEMM for one wave (16 k-rows x 64 b-cols).
// A fragment per ISA 16-bit A 16x32 layout; anchor loaded f32 (non-temporal,
// streamed once through HBM, protecting L2-resident logq) and converted to
// bf16 in registers. Entropy (a*log2 a) fused, striped across the 8 waves so
// each anchor element counts exactly once.
// ---------------------------------------------------------------------------
template <bool TAIL>
__device__ __forceinline__ void kl_step(
    const float* __restrict__ arow,           // anchor + (size_t)k0 * D_DIM
    const unsigned short* __restrict__ lq,
    int it, int lane, int wave, v8f* acc, float& ent)
{
  const int h  = lane >> 4;
  const int r  = lane & 15;
  const int d0 = it * 32;
  const float* p0 = arow + (size_t)r * D_DIM + d0 + 8 * h;  // seg0; seg1 = p0+16

  float af[16];
  if (!TAIL) {
    f32x4 x0 = __builtin_nontemporal_load((const f32x4*)(p0));
    f32x4 x1 = __builtin_nontemporal_load((const f32x4*)(p0 + 4));
    f32x4 y0 = __builtin_nontemporal_load((const f32x4*)(p0 + 16));
    f32x4 y1 = __builtin_nontemporal_load((const f32x4*)(p0 + 20));
#pragma unroll
    for (int j = 0; j < 4; ++j) {
      af[j] = x0[j]; af[4 + j] = x1[j]; af[8 + j] = y0[j]; af[12 + j] = y1[j];
    }
  } else {
#pragma unroll
    for (int j = 0; j < 8; ++j) {
      const int da = d0 + 8 * h + j;
      af[j]     = (da      < D_DIM) ? p0[j]      : 1.0f;  // 1.0: 1*log2(1)=0, pairs with lq=0
      af[8 + j] = (da + 16 < D_DIM) ? p0[16 + j] : 1.0f;
    }
  }

  v16bf a;
#pragma unroll
  for (int j = 0; j < 16; ++j) a[j] = (__bf16)af[j];

  // fused entropy: each d-block's a*log2(a) computed by exactly one of the
  // 8 waves; `wave` is an SGPR (readfirstlane) so this is a scalar branch.
  if ((it & 7) == wave) {
#pragma unroll
    for (int j = 0; j < 16; ++j) ent = __builtin_fmaf(af[j], fast_log2(af[j]), ent);
  }

  // 4 B fragments from L2-resident swizzled log2(q); 32B/lane each (2x b128)
  const char* fbase = (const char*)lq + (((size_t)it * 32 + (wave << 2)) << 10)
                                      + ((size_t)lane << 5);
#pragma unroll
  for (int t = 0; t < 4; ++t) {
    union { u32x4 u[2]; v16bf v; } bf;
    bf.u[0] = *(const u32x4*)(fbase + (size_t)t * 1024);
    bf.u[1] = *(const u32x4*)(fbase + (size_t)t * 1024 + 16);
    acc[t] = __builtin_amdgcn_wmma_f32_16x16x32_bf16(
        false, a, false, bf.v, (short)0, acc[t], false, false);
  }
}

// ---------------------------------------------------------------------------
// Kernel 2: per workgroup = 16 queue rows x all 512 query cols.
// 8 waves x 4 N-tiles. Writes kl2[k][b] = ent2_sum[k] - cross2_sum[k][b]
// (common 1/(D*ln2) factor dropped: rank-invariant for top-k).
// ---------------------------------------------------------------------------
__global__ void __launch_bounds__(256) kl_gemm_kernel(
    const float* __restrict__ anchor, const unsigned short* __restrict__ lq,
    float* __restrict__ klout)
{
  __shared__ float s_ent[16];
  const int tid  = threadIdx.x;
  const int lane = tid & 31;
  const int wave = __builtin_amdgcn_readfirstlane(tid >> 5);  // wave-uniform -> SGPR
  const int k0   = blockIdx.x << 4;
  if (tid < 16) s_ent[tid] = 0.0f;
  __syncthreads();

  const float* arow = anchor + (size_t)k0 * D_DIM;
  v8f   acc[4] = {};
  float ent    = 0.0f;

  for (int it = 0; it < N_DBLK - 1; ++it)
    kl_step<false>(arow, lq, it, lane, wave, acc, ent);
  kl_step<true>(arow, lq, N_DBLK - 1, lane, wave, acc, ent);

  // lanes r and r+16 both hold row r partials -> ds_add_f32 reduce
  atomicAdd(&s_ent[lane & 15], ent);
  __syncthreads();

  // C layout: VGPR r -> M = r + 8*(lane>=16), N = lane&15
  const int h = lane >> 4;
  const int n = lane & 15;
#pragma unroll
  for (int t = 0; t < 4; ++t) {
    const int col = (wave << 6) + (t << 4) + n;
#pragma unroll
    for (int r = 0; r < 8; ++r) {
      const int m = r + (h << 3);
      klout[(size_t)(k0 + m) * B_Q + col] = s_ent[m] - acc[t][r];
    }
  }
}

// ---------------------------------------------------------------------------
// Kernel 3: per query b: each of 128 threads builds a sorted local top-16
// over its strided k slice (register insertion sort), then the global top-16
// is extracted with 16 rounds of a block-wide argmin tree reduction over the
// 128 sorted-list heads (O(1) head access per round since lists are sorted).
// 4-class vote, argmax first-on-tie. kl (16.8 MB) is L2-resident.
// ---------------------------------------------------------------------------
__global__ void __launch_bounds__(128) knn_vote_kernel(
    const float* __restrict__ kl, const int* __restrict__ label,
    float* __restrict__ out)
{
  __shared__ float sv[128 * 16];
  __shared__ int   sl[128 * 16];
  __shared__ float rv[128];
  __shared__ int   rl[128];
  __shared__ int   ri[128];
  const int b   = blockIdx.x;
  const int tid = threadIdx.x;

  float bv[16]; int bl[16];
#pragma unroll
  for (int i = 0; i < 16; ++i) { bv[i] = 3.0e38f; bl[i] = 0; }

  for (int k = tid; k < K_Q; k += 128) {
    const float v = kl[(size_t)k * B_Q + b];
    if (v < bv[15]) {
      bv[15] = v; bl[15] = label[k];
#pragma unroll
      for (int i = 15; i > 0; --i) {
        if (bv[i] < bv[i - 1]) {
          float tv = bv[i]; bv[i] = bv[i - 1]; bv[i - 1] = tv;
          int   tl = bl[i]; bl[i] = bl[i - 1]; bl[i - 1] = tl;
        }
      }
    }
  }

#pragma unroll
  for (int i = 0; i < 16; ++i) { sv[tid * 16 + i] = bv[i]; sl[tid * 16 + i] = bl[i]; }
  __syncthreads();

  int pos = 0;                       // head of this thread's sorted list
  int c0 = 0, c1 = 0, c2 = 0, c3 = 0;
  for (int s = 0; s < 16; ++s) {
    rv[tid] = sv[tid * 16 + pos];
    rl[tid] = sl[tid * 16 + pos];
    ri[tid] = tid;
    __syncthreads();
#pragma unroll
    for (int off = 64; off > 0; off >>= 1) {
      if (tid < off && rv[tid + off] < rv[tid]) {
        rv[tid] = rv[tid + off]; rl[tid] = rl[tid + off]; ri[tid] = ri[tid + off];
      }
      __syncthreads();
    }
    const int wlab = rl[0] & 3;
    const int widx = ri[0];
    __syncthreads();                 // all read winner before next-round writes
    c0 += (wlab == 0); c1 += (wlab == 1); c2 += (wlab == 2); c3 += (wlab == 3);
    if (tid == widx) ++pos;          // winner advances its list head
  }
  if (tid == 0) {
    int bc = 0, bcv = c0;
    if (c1 > bcv) { bc = 1; bcv = c1; }
    if (c2 > bcv) { bc = 2; bcv = c2; }
    if (c3 > bcv) { bc = 3; bcv = c3; }
    out[b] = (float)bc;
  }
}

// ---------------------------------------------------------------------------
extern "C" void kernel_launch(void* const* d_in, const int* in_sizes, int n_in,
                              void* d_out, int out_size, void* d_ws, size_t ws_size,
                              hipStream_t stream) {
  const float* query  = (const float*)d_in[0];   // (512, 50257) f32
  const float* anchor = (const float*)d_in[1];   // (8192, 50257) f32
  const int*   label  = (const int*)d_in[2];     // (8192,) int
  float* out = (float*)d_out;                    // (512,) class index as float

  unsigned short* lq = (unsigned short*)d_ws;                 // 51.5 MB bf16 swizzled log2(q)
  float* klws        = (float*)((char*)d_ws + LQ_BYTES);      // 16.8 MB scores

  dim3 g1((D_PAD + 255) / 256, B_Q);
  logq_swizzle_kernel<<<g1, 256, 0, stream>>>(query, lq);

  kl_gemm_kernel<<<K_Q / 16, 256, 0, stream>>>(anchor, lq, klws);

  knn_vote_kernel<<<B_Q, 128, 0, stream>>>(klws, label, out);
}
